// MultiScalePeriodicPatchEmbedding_57939108823216
// MI455X (gfx1250) — matrix operand order repaired and checked by reading, not compiled, not run on hardware
//
#include <hip/hip_runtime.h>
#include <hip/hip_bf16.h>
#include <math.h>

// ---------------- Problem constants ----------------
#define B_SZ    8
#define E_NUM   30

typedef __attribute__((ext_vector_type(16))) __bf16 v16bf;
typedef __attribute__((ext_vector_type(8)))  float  v8f;
typedef __attribute__((aligned(2))) uint4 uint4_a2;   // possibly 2-byte-aligned 16B load

__constant__ int PATCH_c[E_NUM] = {2,3,4,5,6,7,8,9,10,11,12,13,14,15,16,17,18,19,
                                   21,23,25,28,32,36,42,51,64,85,128,256};
__constant__ int LP_c[E_NUM]    = {128,86,64,52,43,37,32,29,26,24,22,20,19,18,16,16,15,14,
                                   13,12,11,10,8,8,7,6,4,4,2,1};
__constant__ int PPAD_c[E_NUM]  = {32,32,32,32,32,32,32,32,32,32,32,32,32,32,32,32,32,32,
                                   32,32,32,32,32,64,64,64,64,96,128,256};
// element prefix of 256*PPAD
__constant__ int AOFF_c[E_NUM]  = {0,8192,16384,24576,32768,40960,49152,57344,65536,73728,
                                   81920,90112,98304,106496,114688,122880,131072,139264,
                                   147456,155648,163840,172032,180224,188416,204800,221184,
                                   237568,253952,278528,311296};

// ---------------- Workspace layout (bytes); total ~2.32 MB ----------------
#define WS_XT    0u          // 256x256 bf16 x, row-major (b*32+c, t)        (128 KB)
#define WS_XF    131072u     // 256x256 bf16 [xr|xi]                        (128 KB)
#define WS_BTD   262144u     // DFT B, pre-transposed Bt[f][k] 256x256 bf16 (128 KB)
#define WS_BT1   393216u     // MLP1 B, Bt[col][k] 1024x256 bf16            (512 KB)
#define WS_BT2   917504u     // MLP2 B, Bt[col][k] 64x1024 bf16             (128 KB)
#define WS_O1    1048576u    // 256x1024 bf16 [o1r|o1i]                     (512 KB)
#define WS_O2    1572864u    // 256x64 f32                                  (64 KB)
#define WS_META  1638400u    // ints: [0]=counter [1]=total [2..31]=counts [32..271]=rows [272..302]=tilePrefix
#define WS_OOFF  1639936u    // long long outOff[31]
#define WS_BIAS  1640192u    // 30*256 f32
#define WS_ABF   1670912u    // 376832 bf16 A_e (K padded, k-major)         (736 KB)

struct VW { const float* p[E_NUM]; };

// ---------------- WMMA core: 16x16x32 bf16, fragment loads as 16B chunks ----------------
__device__ __forceinline__ v8f wmma_bf16(v16bf a, v16bf b, v8f c) {
  return __builtin_amdgcn_wmma_f32_16x16x32_bf16(false, a, false, b, (short)0, c, false, false);
}

// pA: lane's A row (row m = lane&15), k-major, ld multiple of 16 elements (16B aligned)
// pB: lane's B col (col n = lane&15), k-major, same alignment guarantee
__device__ __forceinline__ v8f wmma_loop(const __bf16* __restrict__ pA,
                                         const __bf16* __restrict__ pB,
                                         int lane, int ksteps, v8f acc) {
  const int abase = (lane < 16) ? 0 : 8;     // A: two runs k=[abase..+7], [abase+16..+23]
  const int kbase = (lane < 16) ? 0 : 16;    // B: same two runs as A, k=[abase..+7], [abase+16..+23]
  for (int kt = 0; kt < ksteps; ++kt) {
    union { v16bf v; uint4 q[2]; } A, Bv;
    const __bf16* a0 = pA + kt * 32 + abase;
    A.q[0] = *(const uint4*)(a0);
    A.q[1] = *(const uint4*)(a0 + 16);
    const __bf16* b0 = pB + kt * 32 + abase;
    Bv.q[0] = *(const uint4*)(b0);
    Bv.q[1] = *(const uint4*)(b0 + 16);
    acc = wmma_bf16(A.v, Bv.v, acc);
  }
  return acc;
}
// C/D fragment: lane holds col n = lane&15; rows m = v + (lane<16 ? 0 : 8)
__device__ __forceinline__ void cCoord(int lane, int v, int& m, int& n) {
  n = lane & 15;
  m = v + ((lane < 16) ? 0 : 8);
}

// ---------------- Fill kernels (one-shot; keep transcendentals OUT of GEMM loops) --------
__global__ __launch_bounds__(256) void prep_x(const float* __restrict__ x, __bf16* __restrict__ Xt) {
  int idx = blockIdx.x * 256 + threadIdx.x;          // 65536
  int n = idx & 255, r = idx >> 8;
  int b = r >> 5, c = r & 31;
  Xt[idx] = (__bf16)x[(b * 256 + n) * 32 + c];
}
__global__ __launch_bounds__(256) void fill_dft(__bf16* __restrict__ BtD) {
  int idx = blockIdx.x * 256 + threadIdx.x;          // BtD[f][k], 256x256
  int f = idx >> 8, k = idx & 255;
  float ang = 6.283185307179586f * (float)((f & 127) + 1) * (float)k * (1.0f / 256.0f);
  BtD[idx] = (__bf16)((f < 128) ? __cosf(ang) : -__sinf(ang));
}
__global__ __launch_bounds__(256) void fill_w1(const float* __restrict__ w1, __bf16* __restrict__ Bt1) {
  int idx = blockIdx.x * 256 + threadIdx.x;          // Bt1[col][k], 1024x256
  int col = idx >> 8, k = idx & 255;
  int j = col & 511; bool im = col >= 512;
  float v;
  if (k < 128) v = im ?  w1[65536 + k * 512 + j] :  w1[k * 512 + j];
  else { int k2 = k - 128;
         v = im ?  w1[k2 * 512 + j] : -w1[65536 + k2 * 512 + j]; }
  Bt1[idx] = (__bf16)v;
}
__global__ __launch_bounds__(256) void fill_w2(const float* __restrict__ w2, __bf16* __restrict__ Bt2) {
  int idx = blockIdx.x * 256 + threadIdx.x;          // Bt2[col][k], 64x1024
  int col = idx >> 10, k = idx & 1023;
  int j = col & 31; bool im = col >= 32;
  float v = 0.0f;
  if (j < 30) {
    if (k < 512) v = im ?  w2[15360 + k * 30 + j] :  w2[k * 30 + j];
    else { int k2 = k - 512;
           v = im ?  w2[k2 * 30 + j] : -w2[15360 + k2 * 30 + j]; }
  }
  Bt2[idx] = (__bf16)v;
}

// ---------------- Gating GEMMs (tight loop: 4x b128 + v_wmma per K-step) ----------------
__global__ __launch_bounds__(32) void rfft_gemm(const __bf16* __restrict__ Xt, const __bf16* __restrict__ BtD,
                                                __bf16* __restrict__ XF) {
  int lane = threadIdx.x, mt = blockIdx.x, nt = blockIdx.y;   // 16 x 16
  v8f acc = {};
  acc = wmma_loop(Xt + (mt * 16 + (lane & 15)) * 256,
                  BtD + (nt * 16 + (lane & 15)) * 256, lane, 8, acc);
#pragma unroll
  for (int v = 0; v < 8; ++v) {
    int m, n; cCoord(lane, v, m, n);
    XF[(mt * 16 + m) * 256 + nt * 16 + n] = (__bf16)acc[v];
  }
}
__global__ __launch_bounds__(32) void mlp1_gemm(const __bf16* __restrict__ XF, const __bf16* __restrict__ Bt1,
                                                const float* __restrict__ b1, __bf16* __restrict__ O1) {
  int lane = threadIdx.x, mt = blockIdx.x, nt = blockIdx.y;   // 16 x 64
  v8f acc = {};
  acc = wmma_loop(XF + (mt * 16 + (lane & 15)) * 256,
                  Bt1 + (nt * 16 + (lane & 15)) * 256, lane, 8, acc);
#pragma unroll
  for (int v = 0; v < 8; ++v) {
    int m, n; cCoord(lane, v, m, n);
    int col = nt * 16 + n;
    O1[(mt * 16 + m) * 1024 + col] = (__bf16)fmaxf(acc[v] + b1[col], 0.0f);
  }
}
__global__ __launch_bounds__(32) void mlp2_gemm(const __bf16* __restrict__ O1, const __bf16* __restrict__ Bt2,
                                                const float* __restrict__ b2, float* __restrict__ O2) {
  int lane = threadIdx.x, mt = blockIdx.x, nt = blockIdx.y;   // 16 x 4
  v8f acc = {};
  acc = wmma_loop(O1 + (mt * 16 + (lane & 15)) * 1024,
                  Bt2 + (nt * 16 + (lane & 15)) * 1024, lane, 32, acc);
#pragma unroll
  for (int v = 0; v < 8; ++v) {
    int m, n; cCoord(lane, v, m, n);
    int col = nt * 16 + n;
    int j = col & 31;
    float bias = (j < 30) ? b2[(col >= 32 ? 30 : 0) + j] : 0.0f;
    O2[(mt * 16 + m) * 64 + col] = acc[v] + bias;
  }
}

// ---------------- Gates + dispatch metadata ----------------
__global__ __launch_bounds__(256) void gates_meta(const float* __restrict__ O2, float* __restrict__ out_gates,
                                                  int* __restrict__ meta, long long* __restrict__ outOff) {
  __shared__ float wv[240];
  __shared__ float g[240];
  int t = threadIdx.x;
  if (t < 240) {
    int b = t / 30, e = t % 30;
    float s = 0.0f;
    for (int c = 0; c < 32; ++c) {
      float orr = O2[(b * 32 + c) * 64 + e];
      float oi  = O2[(b * 32 + c) * 64 + 32 + e];
      s += sqrtf(orr * orr + oi * oi);
    }
    wv[t] = s * (1.0f / 32.0f);
  }
  __syncthreads();
  if (t < 8) {
    float vals[30]; int sel[5]; float tw[5];
    for (int e = 0; e < 30; ++e) vals[e] = wv[t * 30 + e];
    for (int i = 0; i < 5; ++i) {
      int best = 0; float bv = -1e30f;
      for (int e = 0; e < 30; ++e) if (vals[e] > bv) { bv = vals[e]; best = e; }
      sel[i] = best; tw[i] = bv; vals[best] = -1e30f;
    }
    float mx = tw[0], se = 0.0f;
    for (int i = 0; i < 5; ++i) { tw[i] = __expf(tw[i] - mx); se += tw[i]; }
    for (int e = 0; e < 30; ++e) g[t * 30 + e] = 0.0f;
    for (int i = 0; i < 5; ++i) g[t * 30 + sel[i]] = tw[i] / se;
    for (int e = 0; e < 30; ++e) out_gates[t * 30 + e] = g[t * 30 + e];
  }
  __syncthreads();
  if (t == 0) {
    int* counts = meta + 2;
    int* rows   = meta + 32;
    int* tp     = meta + 272;
    long long off = 0; int tacc = 0;
    for (int e = 0; e < 30; ++e) {
      int cnt = 0;
      for (int b = 0; b < 8; ++b) if (g[b * 30 + e] > 0.0f) rows[e * 8 + cnt++] = b;
      counts[e] = cnt;
      tp[e] = tacc;
      outOff[e] = off;
      tacc += cnt * (2 * LP_c[e]) * 16;
      off  += (long long)cnt * 32LL * (long long)LP_c[e] * 256LL;
    }
    tp[30] = tacc; outOff[30] = off;
    meta[1] = tacc;
    meta[0] = 0;
  }
}

// ---------------- A_e reduction: the HBM-bound 257 MB stream ----------------
// lane owns 8 contiguous d' (two float4 loads, 1KB contiguous per wave per j)
__global__ __launch_bounds__(256) void a_reduce(VW vw, const float* __restrict__ sw, const float* __restrict__ sb,
                                               const int* __restrict__ meta, float* __restrict__ bias,
                                               __bf16* __restrict__ Abf) {
  int e = blockIdx.x >> 8;
  int d = blockIdx.x & 255;
  if (meta[2 + e] == 0) return;
  int p = PATCH_c[e], ppad = PPAD_c[e];
  int lane = threadIdx.x & 31, w = threadIdx.x >> 5;
  const float* row = vw.p[e] + (size_t)d * (size_t)(p * 256);
  float swr[8], sbr[8];
#pragma unroll
  for (int q = 0; q < 8; ++q) { swr[q] = sw[lane * 8 + q]; sbr[q] = sb[lane * 8 + q]; }
  __bf16* Arow = Abf + (size_t)AOFF_c[e] + (size_t)d * (size_t)ppad;
  float bsum = 0.0f;
  for (int j = w; j < p; j += 8) {
    const float4* rp = (const float4*)(row + j * 256 + lane * 8);
    __builtin_prefetch((const void*)(rp + 512), 0, 1);   // global_prefetch_b8 ahead
    float4 v0 = rp[0], v1 = rp[1];
    float a, bb;
    a  = v0.x * swr[0]; a  = fmaf(v0.y, swr[1], a); a  = fmaf(v0.z, swr[2], a); a  = fmaf(v0.w, swr[3], a);
    a  = fmaf(v1.x, swr[4], a); a = fmaf(v1.y, swr[5], a); a = fmaf(v1.z, swr[6], a); a = fmaf(v1.w, swr[7], a);
    bb = v0.x * sbr[0]; bb = fmaf(v0.y, sbr[1], bb); bb = fmaf(v0.z, sbr[2], bb); bb = fmaf(v0.w, sbr[3], bb);
    bb = fmaf(v1.x, sbr[4], bb); bb = fmaf(v1.y, sbr[5], bb); bb = fmaf(v1.z, sbr[6], bb); bb = fmaf(v1.w, sbr[7], bb);
#pragma unroll
    for (int s = 16; s > 0; s >>= 1) { a += __shfl_xor(a, s, 32); bb += __shfl_xor(bb, s, 32); }
    if (lane == 0) Arow[j] = (__bf16)a;
    bsum += bb;
  }
  if (lane == 0) for (int j = p + w; j < ppad; j += 8) Arow[j] = (__bf16)0.0f;
  __shared__ float bs[8];
  if (lane == 0) bs[w] = bsum;
  __syncthreads();
  if (threadIdx.x == 0) {
    float s = 0.0f;
    for (int i = 0; i < 8; ++i) s += bs[i];
    bias[e * 256 + d] = s;
  }
}

// ---------------- Persistent patch-embed WMMA GEMM ----------------
__device__ __forceinline__ uint4 loadXp8(const __bf16* __restrict__ Xrow, int tbase, int jstart, int p) {
  if ((jstart + 7 < p) && (tbase + jstart + 7 <= 255))
    return *(const uint4_a2*)(Xrow + tbase + jstart);  // contiguous fast path
  union { uint4 q; __bf16 h[8]; } u;
#pragma unroll
  for (int i = 0; i < 8; ++i) {
    int j = jstart + i;
    __bf16 v = (__bf16)0.0f;
    if (j < p) { int tc = tbase + j; if (tc > 255) tc = 255; v = Xrow[tc]; }  // edge pad
    u.h[i] = v;
  }
  return u.q;
}

__global__ __launch_bounds__(32) void patch_gemm(const __bf16* __restrict__ Xt, const __bf16* __restrict__ Abf,
                                                 const float* __restrict__ bias, int* __restrict__ meta,
                                                 const long long* __restrict__ outOff, float* __restrict__ out) {
  int lane = threadIdx.x;
  int total = meta[1];
  const int* tp = meta + 272;
  const int abase = (lane < 16) ? 0 : 8;
  const int kbase = (lane < 16) ? 0 : 16;
  for (;;) {
    int t = 0;
    if (lane == 0) t = atomicAdd(meta, 1);
    t = __builtin_amdgcn_readfirstlane(t);
    if (t >= total) break;
    int e = 0;
    while (e < 29 && t >= tp[e + 1]) ++e;
    int local = t - tp[e];
    int Lp = LP_c[e], p = PATCH_c[e], ppad = PPAD_c[e];
    int tilesPerB = 2 * Lp * 16;
    int bslot = local / tilesPerB;
    int rem = local - bslot * tilesPerB;
    int mt = rem >> 4, nt = rem & 15;
    int b = meta[32 + e * 8 + bslot];
    // lane-invariant A row (m = lane&15) and B col (n = lane&15) pointers
    int mrow = lane & 15;
    int R = mt * 16 + mrow;
    int c = R / Lp, l = R - c * Lp;
    const __bf16* Xrow = Xt + (b * 32 + c) * 256;
    int tbase = l * p;
    const __bf16* pB = Abf + (size_t)AOFF_c[e] + (size_t)(nt * 16 + mrow) * (size_t)ppad;
    v8f acc = {};
    for (int kt = 0; kt < (ppad >> 5); ++kt) {
      union { v16bf v; uint4 q[2]; } A, Bv;
      A.q[0] = loadXp8(Xrow, tbase, kt * 32 + abase, p);
      A.q[1] = loadXp8(Xrow, tbase, kt * 32 + abase + 16, p);
      const __bf16* b0 = pB + kt * 32 + abase;
      Bv.q[0] = *(const uint4*)(b0);
      Bv.q[1] = *(const uint4*)(b0 + 16);
      acc = wmma_bf16(A.v, Bv.v, acc);
    }
    long long base = outOff[e];
    // C fragment: column d fixed per lane -> hoist pos-emb frequency + bias
    int d = nt * 16 + (lane & 15);
    float dv = __expf((float)(d & ~1) * (-0.03597789207803197f));  // -ln(10000)/256
    float bval = bias[e * 256 + d];
    bool odd = (d & 1);
#pragma unroll
    for (int v = 0; v < 8; ++v) {
      int m = v + ((lane < 16) ? 0 : 8);
      int Rv = mt * 16 + m;
      int cv = Rv / Lp, lv = Rv - cv * Lp;
      float ang = (float)lv * dv;
      float pe = odd ? __cosf(ang) : __sinf(ang);
      long long idx = base + ((long long)(bslot * 32 + cv) * Lp + lv) * 256 + d;
      out[idx] = acc[v] + bval + pe;
    }
  }
}

// ---------------- Launch ----------------
extern "C" void kernel_launch(void* const* d_in, const int* in_sizes, int n_in,
                              void* d_out, int out_size, void* d_ws, size_t ws_size,
                              hipStream_t stream) {
  (void)in_sizes; (void)n_in; (void)ws_size;   // needs ~2.32 MB of d_ws
  const float* x  = (const float*)d_in[0];
  const float* w1 = (const float*)d_in[1];
  const float* b1 = (const float*)d_in[2];
  const float* w2 = (const float*)d_in[3];
  const float* b2 = (const float*)d_in[4];
  const float* sw = (const float*)d_in[5];
  const float* sb = (const float*)d_in[6];
  VW vw;
  for (int i = 0; i < E_NUM; ++i) vw.p[i] = (const float*)d_in[7 + i];

  char* ws = (char*)d_ws;
  __bf16*    Xt    = (__bf16*)(ws + WS_XT);
  __bf16*    XF    = (__bf16*)(ws + WS_XF);
  __bf16*    BtD   = (__bf16*)(ws + WS_BTD);
  __bf16*    Bt1   = (__bf16*)(ws + WS_BT1);
  __bf16*    Bt2   = (__bf16*)(ws + WS_BT2);
  __bf16*    O1    = (__bf16*)(ws + WS_O1);
  float*     O2    = (float*)(ws + WS_O2);
  int*       meta  = (int*)(ws + WS_META);
  long long* oOff  = (long long*)(ws + WS_OOFF);
  float*     biasb = (float*)(ws + WS_BIAS);
  __bf16*    Abf   = (__bf16*)(ws + WS_ABF);

  float* out       = (float*)d_out;
  float* out_gates = out + (out_size - B_SZ * E_NUM);

  prep_x  <<<256,  256, 0, stream>>>(x, Xt);
  fill_dft<<<256,  256, 0, stream>>>(BtD);
  fill_w1 <<<1024, 256, 0, stream>>>(w1, Bt1);
  fill_w2 <<<256,  256, 0, stream>>>(w2, Bt2);
  rfft_gemm<<<dim3(16, 16), 32, 0, stream>>>(Xt, BtD, XF);
  mlp1_gemm<<<dim3(16, 64), 32, 0, stream>>>(XF, Bt1, b1, O1);
  mlp2_gemm<<<dim3(16, 4),  32, 0, stream>>>(O1, Bt2, b2, O2);
  gates_meta<<<1, 256, 0, stream>>>(O2, out_gates, meta, oOff);
  a_reduce <<<E_NUM * 256, 256, 0, stream>>>(vw, sw, sb, meta, biasb, Abf);
  patch_gemm<<<1024, 32, 0, stream>>>(Xt, Abf, biasb, meta, oOff, out);
}
